// NearestBasisInteraction_65962107732488
// MI455X (gfx1250) — matrix operand order repaired and checked by reading, not compiled, not run on hardware
//
#include <hip/hip_runtime.h>
#include <hip/hip_bf16.h>

typedef __attribute__((ext_vector_type(16))) _Float16 v16h;
typedef __attribute__((ext_vector_type(8)))  float    v8f;
typedef unsigned int u32;

union FragA {
    v16h h;
    uint4 q[2];
    _Float16 f[16];
    u32 u[8];
};

__device__ __forceinline__ float silu_f(float x) {
    return x / (1.0f + __expf(-x));
}

__device__ __forceinline__ v8f wmma16(const FragA& a, const FragA& b, v8f c) {
    // D = A(16x32 f16) x B(32x16 f16) + C(16x16 f32)
    return __builtin_amdgcn_wmma_f32_16x16x32_f16(
        /*neg_a=*/false, a.h, /*neg_b=*/false, b.h,
        /*c_mod=*/(short)0, c, /*reuse_a=*/false, /*reuse_b=*/false);
}

// ---------------------------------------------------------------------------
// Pack a (K x N) row-major f32 weight into WMMA B-fragment order:
// dst[kc][no][lane][v] (dwords), dword v of lane holds f16 pair
//   K = kc*32 + 16*(v>=4) + 8*(lane>=16) + 2*(v&3) + {0, 1},  col = no*16 + (lane&15)
// so each wave's B fragment is 2 contiguous uint4 loads per lane.
// ---------------------------------------------------------------------------
__device__ void pack_one(const float* W, int K, int N, u32* dst, int tid, int nth) {
    int ntiles = N >> 4;
    int total = (K >> 5) * ntiles * 256;  // dwords
    for (int i = tid; i < total; i += nth) {
        int v    = i & 7;
        int lane = (i >> 3) & 31;
        int no   = (i >> 8) % ntiles;
        int kc   = i / (256 * ntiles);
        int k0   = (kc << 5) + ((v >> 2) << 4) + ((lane >> 4) << 3) + ((v & 3) << 1);
        int col  = (no << 4) + (lane & 15);
        union { _Float16 f[2]; u32 u; } p;
        p.f[0] = (_Float16)W[(size_t)k0 * N + col];
        p.f[1] = (_Float16)W[(size_t)(k0 + 1) * N + col];
        dst[i] = p.u;
    }
}

__global__ void k_pack_weights(const float* Wd, const float* Wb, const float* Wm,
                               const float* Wst, const float* Wts,
                               u32* WdP, u32* WbP, u32* WmP, u32* WstP, u32* WtsP) {
    int tid = blockIdx.x * blockDim.x + threadIdx.x;
    int nth = gridDim.x * blockDim.x;
    pack_one(Wd,  128,  32, WdP,  tid, nth);
    pack_one(Wb,  1024, 32, WbP,  tid, nth);   // W_bilin flattened (q*32+s, o)
    pack_one(Wm,  64,   32, WmP,  tid, nth);
    pack_one(Wst, 32,  128, WstP, tid, nth);
    pack_one(Wts, 32,  128, WtsP, tid, nth);
}

__global__ void k_zero(float* p, int n) {
    for (int i = blockIdx.x * blockDim.x + threadIdx.x; i < n;
         i += gridDim.x * blockDim.x)
        p[i] = 0.0f;
}

// ---------------------------------------------------------------------------
// h_down = silu(h @ W_down) -> f16.  One wave per 16-atom tile.
// ---------------------------------------------------------------------------
__global__ void k_down(const float* __restrict__ h, const u32* __restrict__ WdP,
                       _Float16* __restrict__ hd16, int N) {
    int wave = (blockIdx.x * blockDim.x + threadIdx.x) >> 5;
    int lane = threadIdx.x & 31;
    int tiles = (N + 15) >> 4;
    if (wave >= tiles) return;
    int m0 = wave << 4;
    int l15 = lane & 15, lhalf = lane >> 4;
    int row = m0 + l15;
    if (row >= N) row = N - 1;
    int khalf = lhalf * 8;

    v8f acc0 = {}, acc1 = {};
    for (int kc = 0; kc < 4; ++kc) {
        FragA a;
        const float* src = h + (size_t)row * 128 + kc * 32 + khalf;
#pragma unroll
        for (int j = 0; j < 8; ++j) a.f[j] = (_Float16)src[j];
#pragma unroll
        for (int j = 0; j < 8; ++j) a.f[8 + j] = (_Float16)src[16 + j];
        FragA b0, b1;
        const u32* p0 = WdP + ((size_t)(kc * 2 + 0) * 32 + lane) * 8;
        const u32* p1 = WdP + ((size_t)(kc * 2 + 1) * 32 + lane) * 8;
        b0.q[0] = *(const uint4*)p0;       b0.q[1] = *(const uint4*)(p0 + 4);
        b1.q[0] = *(const uint4*)p1;       b1.q[1] = *(const uint4*)(p1 + 4);
        acc0 = wmma16(a, b0, acc0);
        acc1 = wmma16(a, b1, acc1);
    }
    int mbase = m0 + lhalf * 8;
#pragma unroll
    for (int r = 0; r < 8; ++r) {
        int mrow = mbase + r;
        if (mrow < N) {
            hd16[(size_t)mrow * 32 + l15]      = (_Float16)silu_f(acc0[r]);
            hd16[(size_t)mrow * 32 + 16 + l15] = (_Float16)silu_f(acc1[r]);
        }
    }
}

// ---------------------------------------------------------------------------
// Interaction: per block = 16 edges.
//   P[e] = m2[e]^T @ (rbf*sbf)[e]  (32x8x32, K-padded WMMA)
//   h_sbf_e = P_flat @ W_bilin     ((16x1024)@(1024x32) WMMA GEMM)
//   LDS reduction of K-partials, then atomic scatter-add into h_mp[idx_s]
// ---------------------------------------------------------------------------
#define TE 16
__global__ __launch_bounds__(256) void k_interact(
    const float* __restrict__ rbf, const float* __restrict__ sbf,
    const _Float16* __restrict__ hd16, const u32* __restrict__ WbP,
    const int* __restrict__ idx_s, const int* __restrict__ idx_t,
    const int* __restrict__ enb, float* __restrict__ h_mp, int E) {

    // stage area (16 KB) is dead after the P phase -> reuse it for the
    // bilinear partial-sum reduction (8 KB) to cut global atomics 4x.
    __shared__ union {
        struct {
            _Float16 mT[TE][32][8];   // [e][q][k]  8 KB
            _Float16 bT[TE][32][8];   // [e][s][k]  8 KB
        } stg;
        float accS[2][4][256];        // [ntile][kquarter][m*16+n] 8 KB
    } sh;
    __shared__ _Float16 Pl[TE][1024]; // [e][q*32+s] 32 KB (f16)

    int tid = threadIdx.x;
    int eBase = blockIdx.x * TE;

    // ---- stage m^T: for each (e,q) fill k=0..7 contiguously
    for (int slot = tid; slot < TE * 32; slot += 256) {
        int e = slot >> 5, q = slot & 31;
        int eg = eBase + e;
        if (eg < E) {
#pragma unroll
            for (int k = 0; k < 8; ++k) {
                int j  = eg * 8 + k;
                int ei = enb[j];                 // edge owning nb entry j
                sh.stg.mT[e][q][k] = hd16[(size_t)idx_t[ei] * 32 + q];
            }
        } else {
#pragma unroll
            for (int k = 0; k < 8; ++k) sh.stg.mT[e][q][k] = (_Float16)0.0f;
        }
    }
    // ---- stage basis^T: row (e,k), 32 s-lanes coalesced on rbf/sbf
    {
        int s = tid & 31;
        for (int rr = tid >> 5; rr < TE * 8; rr += 8) {
            int e = rr >> 3, k = rr & 7;
            int eg = eBase + e;
            float v = 0.0f;
            if (eg < E) {
                size_t j = (size_t)eg * 8 + k;
                v = rbf[j * 32 + s] * sbf[j * 32 + s];
                // prefetch next block's rows of the dominant 307 MB streams
                if (eg + TE < E) {
                    __builtin_prefetch(rbf + (j + TE * 8) * 32 + s, 0, 1);
                    __builtin_prefetch(sbf + (j + TE * 8) * 32 + s, 0, 1);
                }
            }
            sh.stg.bT[e][s][k] = (_Float16)v;
        }
    }
    __syncthreads();

    int wave = tid >> 5, lane = tid & 31;
    int lhalf = lane >> 4, l15 = lane & 15;
    const uint4 z4 = {0u, 0u, 0u, 0u};

    // ---- P phase: each wave computes 2 edges' 32x32 P (K=8 padded to 32)
    for (int ee = 0; ee < 2; ++ee) {
        int e = wave * 2 + ee;
        FragA am[2], bb[2];
#pragma unroll
        for (int mq = 0; mq < 2; ++mq) {
            am[mq].q[0] = (lhalf == 0) ? *(const uint4*)&sh.stg.mT[e][mq * 16 + l15][0] : z4;
            am[mq].q[1] = z4;   // K=8..31 zero pad
        }
#pragma unroll
        for (int ns = 0; ns < 2; ++ns) {
            bb[ns].q[0] = (lhalf == 0) ? *(const uint4*)&sh.stg.bT[e][ns * 16 + l15][0] : z4;
            bb[ns].q[1] = z4;
        }
#pragma unroll
        for (int mq = 0; mq < 2; ++mq) {
#pragma unroll
            for (int ns = 0; ns < 2; ++ns) {
                v8f acc = {};
                acc = wmma16(am[mq], bb[ns], acc);
                int qb = mq * 16 + lhalf * 8;
                int sb = ns * 16 + l15;
#pragma unroll
                for (int r = 0; r < 8; ++r)
                    Pl[e][(qb + r) * 32 + sb] = (_Float16)acc[r];
            }
        }
    }
    __syncthreads();   // Pl ready; stage area no longer read after this point

    // ---- bilinear GEMM: wave -> (N-tile = wave&1, K-quarter = wave>>1)
    int no  = wave & 1;
    int kq  = wave >> 1;
    int kc0 = kq * 8;
    int koff = lhalf * 8;
    v8f acc = {};
    for (int kc = kc0; kc < kc0 + 8; ++kc) {
        FragA a, b;
        a.q[0] = *(const uint4*)&Pl[l15][kc * 32 + koff];
        a.q[1] = *(const uint4*)&Pl[l15][kc * 32 + 16 + koff];
        const u32* bp = WbP + ((size_t)(kc * 2 + no) * 32 + lane) * 8;
        b.q[0] = *(const uint4*)bp;
        b.q[1] = *(const uint4*)(bp + 4);
        acc = wmma16(a, b, acc);
    }
    // ---- store K-quarter partials to LDS (reuses dead stage memory)
#pragma unroll
    for (int r = 0; r < 8; ++r)
        sh.accS[no][kq][(r + 8 * lhalf) * 16 + l15] = acc[r];
    __syncthreads();

    // ---- waves 0,1 reduce the 4 partials and issue the 512 unique atomics
    if (wave < 2) {
        int nn = wave;  // output N-tile
#pragma unroll
        for (int r = 0; r < 8; ++r) {
            int m = r + 8 * lhalf;
            float v = sh.accS[nn][0][m * 16 + l15] + sh.accS[nn][1][m * 16 + l15] +
                      sh.accS[nn][2][m * 16 + l15] + sh.accS[nn][3][m * 16 + l15];
            int e = eBase + m;
            if (e < E)
                atomicAdd(&h_mp[(size_t)idx_s[e] * 32 + nn * 16 + l15], v);
        }
    }
}

__global__ void k_hmp(const float* __restrict__ h_mp, const float* __restrict__ scale,
                      _Float16* __restrict__ hmp16, int n) {
    float sc = scale[0];
    for (int i = blockIdx.x * blockDim.x + threadIdx.x; i < n;
         i += gridDim.x * blockDim.x)
        hmp16[i] = (_Float16)(h_mp[i] * sc);
}

// ---------------------------------------------------------------------------
// x = silu(concat(h_mp[idx_s], h_mp[idx_t]) @ W_mst) -> x16 (E x 32, f16)
// ---------------------------------------------------------------------------
__global__ void k_mst(const _Float16* __restrict__ hmp16, const u32* __restrict__ WmP,
                      const int* __restrict__ idx_s, const int* __restrict__ idx_t,
                      _Float16* __restrict__ x16, int E) {
    int wave = (blockIdx.x * blockDim.x + threadIdx.x) >> 5;
    int lane = threadIdx.x & 31;
    int tiles = (E + 15) >> 4;
    if (wave >= tiles) return;
    int e0 = wave << 4;
    int lhalf = lane >> 4, l15 = lane & 15;
    int erow = e0 + l15;
    bool ok = erow < E;
    int rs = ok ? idx_s[erow] : 0;
    int rt = ok ? idx_t[erow] : 0;
    int koff = lhalf * 8;
    const uint4 z4 = {0u, 0u, 0u, 0u};

    v8f acc0 = {}, acc1 = {};
    for (int kc = 0; kc < 2; ++kc) {
        const _Float16* src = hmp16 + (size_t)(kc == 0 ? rs : rt) * 32;
        FragA a;
        a.q[0] = ok ? *(const uint4*)(src + koff)      : z4;
        a.q[1] = ok ? *(const uint4*)(src + koff + 16) : z4;
        FragA b0, b1;
        const u32* p0 = WmP + ((size_t)(kc * 2 + 0) * 32 + lane) * 8;
        const u32* p1 = WmP + ((size_t)(kc * 2 + 1) * 32 + lane) * 8;
        b0.q[0] = *(const uint4*)p0;  b0.q[1] = *(const uint4*)(p0 + 4);
        b1.q[0] = *(const uint4*)p1;  b1.q[1] = *(const uint4*)(p1 + 4);
        acc0 = wmma16(a, b0, acc0);
        acc1 = wmma16(a, b1, acc1);
    }
    int mb = e0 + lhalf * 8;
#pragma unroll
    for (int r = 0; r < 8; ++r) {
        int e = mb + r;
        if (e < E) {
            x16[(size_t)e * 32 + l15]      = (_Float16)silu_f(acc0[r]);
            x16[(size_t)e * 32 + 16 + l15] = (_Float16)silu_f(acc1[r]);
        }
    }
}

// ---------------------------------------------------------------------------
// out[j] = (silu(x[j]@W_st) + silu(x[idx_swap[j]]@W_ts)) / sqrt(2)
// Fuses the reverse-edge gather into the A-fragment load.
// ---------------------------------------------------------------------------
__global__ void k_out(const _Float16* __restrict__ x16, const u32* __restrict__ WstP,
                      const u32* __restrict__ WtsP, const int* __restrict__ idx_sw,
                      float* __restrict__ out, int E) {
    int wave = (blockIdx.x * blockDim.x + threadIdx.x) >> 5;
    int lane = threadIdx.x & 31;
    int tiles = (E + 15) >> 4;
    if (wave >= tiles) return;
    int e0 = wave << 4;
    int lhalf = lane >> 4, l15 = lane & 15;
    int koff = lhalf * 8;
    int jrow = e0 + l15;
    bool ok = jrow < E;
    int srow = ok ? idx_sw[jrow] : 0;
    const uint4 z4 = {0u, 0u, 0u, 0u};

    FragA ast, ats;
    ast.q[0] = ok ? *(const uint4*)(x16 + (size_t)jrow * 32 + koff)      : z4;
    ast.q[1] = ok ? *(const uint4*)(x16 + (size_t)jrow * 32 + koff + 16) : z4;
    ats.q[0] = ok ? *(const uint4*)(x16 + (size_t)srow * 32 + koff)      : z4;
    ats.q[1] = ok ? *(const uint4*)(x16 + (size_t)srow * 32 + koff + 16) : z4;

    const float c = 0.70710678118654752440f;
    int mb = e0 + lhalf * 8;
    for (int no = 0; no < 8; ++no) {
        FragA bst, bts;
        const u32* ps = WstP + ((size_t)no * 32 + lane) * 8;
        const u32* pt = WtsP + ((size_t)no * 32 + lane) * 8;
        bst.q[0] = *(const uint4*)ps;  bst.q[1] = *(const uint4*)(ps + 4);
        bts.q[0] = *(const uint4*)pt;  bts.q[1] = *(const uint4*)(pt + 4);
        v8f a_s = {}, a_t = {};
        a_s = wmma16(ast, bst, a_s);
        a_t = wmma16(ats, bts, a_t);
#pragma unroll
        for (int r = 0; r < 8; ++r) {
            int e = mb + r;
            if (e < E)
                out[(size_t)e * 128 + no * 16 + l15] =
                    (silu_f(a_s[r]) + silu_f(a_t[r])) * c;
        }
    }
}

// ---------------------------------------------------------------------------
extern "C" void kernel_launch(void* const* d_in, const int* in_sizes, int n_in,
                              void* d_out, int out_size, void* d_ws, size_t ws_size,
                              hipStream_t stream) {
    const float* h     = (const float*)d_in[0];
    const float* rbf   = (const float*)d_in[1];
    const float* sbf   = (const float*)d_in[2];
    const float* Wd    = (const float*)d_in[3];
    const float* Wb    = (const float*)d_in[4];
    const float* Wm    = (const float*)d_in[5];
    const float* Wst   = (const float*)d_in[6];
    const float* Wts   = (const float*)d_in[7];
    const float* scale = (const float*)d_in[8];
    const int* idx_s   = (const int*)d_in[9];
    const int* idx_t   = (const int*)d_in[10];
    const int* idx_sw  = (const int*)d_in[11];
    const int* enb     = (const int*)d_in[12];
    // d_in[13] (edge_nb_ragged_idx) == tile(arange(K)) by construction

    int N = in_sizes[0] / 128;
    int E = in_sizes[9];
    float* out = (float*)d_out;

    size_t off = 0;
    auto wsalloc = [&](size_t bytes) -> void* {
        void* p = (char*)d_ws + off;
        off += (bytes + 255) & ~(size_t)255;
        return p;
    };
    u32* WdP  = (u32*)wsalloc(4 * 2 * 32 * 8 * 4);
    u32* WbP  = (u32*)wsalloc(32 * 2 * 32 * 8 * 4);
    u32* WmP  = (u32*)wsalloc(2 * 2 * 32 * 8 * 4);
    u32* WstP = (u32*)wsalloc(8 * 32 * 8 * 4);
    u32* WtsP = (u32*)wsalloc(8 * 32 * 8 * 4);
    _Float16* hd16  = (_Float16*)wsalloc((size_t)N * 32 * 2);
    float*    h_mp  = (float*)wsalloc((size_t)N * 32 * 4);
    _Float16* hmp16 = (_Float16*)wsalloc((size_t)N * 32 * 2);
    _Float16* x16   = (_Float16*)wsalloc((size_t)E * 32 * 2);

    k_pack_weights<<<64, 256, 0, stream>>>(Wd, Wb, Wm, Wst, Wts,
                                           WdP, WbP, WmP, WstP, WtsP);
    k_zero<<<512, 256, 0, stream>>>(h_mp, N * 32);

    int tilesN = (N + 15) >> 4;
    k_down<<<(tilesN + 3) / 4, 128, 0, stream>>>(h, WdP, hd16, N);

    k_interact<<<(E + TE - 1) / TE, 256, 0, stream>>>(rbf, sbf, hd16, WbP,
                                                      idx_s, idx_t, enb, h_mp, E);

    k_hmp<<<512, 256, 0, stream>>>(h_mp, scale, hmp16, N * 32);

    int tilesE = (E + 15) >> 4;
    k_mst<<<(tilesE + 3) / 4, 128, 0, stream>>>(hmp16, WmP, idx_s, idx_t, x16, E);
    k_out<<<(tilesE + 3) / 4, 128, 0, stream>>>(x16, WstP, WtsP, idx_sw, out, E);
}
